// WindowMSA_12335146074706
// MI455X (gfx1250) — matrix operand order repaired and checked by reading, not compiled, not run on hardware
//
#include <hip/hip_runtime.h>
#include <math.h>

#define B_WIN 1024
#define N_TOK 64
#define DIM   128
#define HEADS 4
#define HD    384
#define HDT   1536
#define LDH   136   // f16 k-stride for 128-wide tiles (bank-conflict-free b128)
#define LDV   72    // f16 k-stride for 64-wide tiles (Ps, Vt)
#define LOG100 4.605170185988092f

typedef _Float16 h16;
typedef __attribute__((ext_vector_type(16))) _Float16 v16h;
typedef __attribute__((ext_vector_type(8)))  float    v8f;

__device__ __forceinline__ v8f vzero8() {
  v8f z;
  for (int i = 0; i < 8; ++i) z[i] = 0.f;
  return z;
}

__device__ __forceinline__ v8f wmma16(v16h a, v16h b, v8f c) {
  return __builtin_amdgcn_wmma_f32_16x16x32_f16(false, a, false, b, (short)0, c,
                                                false, false);
}

// A fragment (16x32 f16): lane holds K = kBase + lhi*8 + {0..7} and +16.
__device__ __forceinline__ v16h afrag(const h16* s, int ld, int row, int kBase, int lhi) {
  const h16* p = s + row * ld + kBase + lhi * 8;
  union { v16h v; uint4 q[2]; } x;
  x.q[0] = *(const uint4*)p;
  x.q[1] = *(const uint4*)(p + 16);
  return x.v;
}

// B fragment (32x16 f16) read from transposed storage Bt[n][k]:
// lane holds K = kBase + lhi*8 + {0..7} and +16.
__device__ __forceinline__ v16h bfrag(const h16* s, int ld, int row, int kBase, int lhi) {
  const h16* p = s + row * ld + kBase + lhi * 8;
  union { v16h v; uint4 q[2]; } x;
  x.q[0] = *(const uint4*)p;
  x.q[1] = *(const uint4*)(p + 16);
  return x.v;
}

// ---------------- one-time weight convert + transpose (fp32 -> f16) ----------------
// Wt[n][k] = (f16) W[k][n]
__global__ void transpose_conv_kernel(const float* __restrict__ W, h16* __restrict__ Wt,
                                      int K, int N) {
  __shared__ float tile[32][33];
  int nb = blockIdx.x * 32, kb = blockIdx.y * 32;
  int tx = threadIdx.x & 31, ty = threadIdx.x >> 5;
  for (int r = ty; r < 32; r += 8)
    tile[r][tx] = W[(size_t)(kb + r) * N + nb + tx];
  __syncthreads();
  for (int r = ty; r < 32; r += 8)
    Wt[(size_t)(nb + r) * K + kb + tx] = (h16)tile[tx][r];
}

// ---------------- bias precompute (data-independent) ----------------
__global__ void bias_mlp_kernel(const float* __restrict__ relpos, const float* __restrict__ w1,
                                const float* __restrict__ b1, const float* __restrict__ w2,
                                float* __restrict__ table) {
  int g = blockIdx.x * blockDim.x + threadIdx.x;
  if (g >= 225 * 4) return;
  int i = g >> 2, h = g & 3;
  float x0 = relpos[i * 2], x1 = relpos[i * 2 + 1];
  float s = 0.f;
  for (int j = 0; j < 512; ++j) {
    float r = fmaxf(x0 * w1[j] + x1 * w1[512 + j] + b1[j], 0.f);
    s += r * w2[j * 4 + h];
  }
  table[g] = s;
}

__global__ void bias_expand_kernel(const int* __restrict__ relidx,
                                   const float* __restrict__ table,
                                   float* __restrict__ Bias) {
  int g = blockIdx.x * blockDim.x + threadIdx.x;
  if (g >= 4 * 64 * 64) return;
  int h = g >> 12, p = (g >> 6) & 63, q = g & 63;
  int n = ((p >> 3) << 3) + (q >> 3);
  int m = ((p & 7) << 3) + (q & 7);
  float x = table[relidx[n * 64 + m] * 4 + h];
  Bias[g] = 16.f / (1.f + expf(-x));
}

// ---------------- fused window-MSA kernel ----------------
__global__ __launch_bounds__(256) void winmsa_kernel(
    const float* __restrict__ E,
    const h16* __restrict__ WQt, const h16* __restrict__ WKt,
    const h16* __restrict__ WVt, const h16* __restrict__ WAt,
    const float* __restrict__ WQb, const float* __restrict__ WVb,
    const float* __restrict__ WAb, const float* __restrict__ lscale,
    const float* __restrict__ Bias, float* __restrict__ out) {
  __shared__ h16 Esh[N_TOK * LDH];   // E tile (f16)      [tok][k=128]
  __shared__ h16 Qb[N_TOK * LDH];    // Q chunk           [tok][k=128]
  __shared__ h16 Kb[N_TOK * LDH];    // K chunk           [tok][k=128]
  __shared__ h16 Ac[N_TOK * LDH];    // A chunk           [tok][k=128]
  __shared__ h16 Wt[DIM * LDH];      // staged weights    [col][k=128]
  __shared__ h16 Vt[DIM * LDV];      // V chunk transposed[col][tok=64]
  __shared__ h16 Ps[N_TOK * LDV];    // attention         [tok][tok=64]
  __shared__ float qss[N_TOK];
  __shared__ float kss[N_TOK];

  const int tid  = threadIdx.x;
  const int w    = tid >> 5;
  const int lane = tid & 31;
  const int lhi  = lane >> 4;
  const int llo  = lane & 15;
  const int mt   = w >> 1;      // M tile (0..3)
  const int half = w & 1;       // N half
  const int bwin = blockIdx.x;
  const int arow = mt * 16 + llo;

  // stage E (64x128 fp32 -> f16)
  {
    const float4* src = (const float4*)(E + (size_t)bwin * (N_TOK * DIM));
    for (int i = tid; i < 2048; i += 256) {
      float4 v = src[i];
      int r = (i << 2) >> 7, c = (i << 2) & 127;
      h16* d = &Esh[r * LDH + c];
      d[0] = (h16)v.x; d[1] = (h16)v.y; d[2] = (h16)v.z; d[3] = (h16)v.w;
    }
  }

  v8f oacc[4];
  for (int i = 0; i < 4; ++i) oacc[i] = vzero8();

  for (int h = 0; h < HEADS; ++h) {
    v8f p[2];
    p[0] = vzero8(); p[1] = vzero8();

    // ---- build P over 3 chunks of HD ----
    for (int cc = 0; cc < 3; ++cc) {
      const int colBase = h * HD + cc * 128;
      __syncthreads();
      // stage WQ chunk (contiguous 32KB)
      {
        const uint4* s = (const uint4*)(WQt + (size_t)colBase * DIM);
        for (int v = tid; v < 2048; v += 256) {
          int r = v >> 4, kc = (v & 15) << 3;
          *(uint4*)&Wt[r * LDH + kc] = s[v];
        }
      }
      __syncthreads();
      {  // Q chunk = Es @ WQ + bq
        v8f acc[4];
        for (int i = 0; i < 4; ++i) acc[i] = vzero8();
        for (int ks = 0; ks < 4; ++ks) {
          int kb = ks * 32;
          v16h a = afrag(Esh, LDH, arow, kb, lhi);
#pragma unroll
          for (int i = 0; i < 4; ++i) {
            int ncol = (half * 4 + i) * 16 + llo;
            acc[i] = wmma16(a, bfrag(Wt, LDH, ncol, kb, lhi), acc[i]);
          }
        }
#pragma unroll
        for (int i = 0; i < 4; ++i) {
          int col = (half * 4 + i) * 16 + llo;
          float qbias = WQb[colBase + col];
#pragma unroll
          for (int r = 0; r < 8; ++r)
            Qb[(mt * 16 + r + 8 * lhi) * LDH + col] = (h16)(acc[i][r] + qbias);
        }
      }
      __syncthreads();
      // stage WK chunk
      {
        const uint4* s = (const uint4*)(WKt + (size_t)colBase * DIM);
        for (int v = tid; v < 2048; v += 256) {
          int r = v >> 4, kc = (v & 15) << 3;
          *(uint4*)&Wt[r * LDH + kc] = s[v];
        }
      }
      __syncthreads();
      {  // K chunk = Es @ WK
        v8f acc[4];
        for (int i = 0; i < 4; ++i) acc[i] = vzero8();
        for (int ks = 0; ks < 4; ++ks) {
          int kb = ks * 32;
          v16h a = afrag(Esh, LDH, arow, kb, lhi);
#pragma unroll
          for (int i = 0; i < 4; ++i) {
            int ncol = (half * 4 + i) * 16 + llo;
            acc[i] = wmma16(a, bfrag(Wt, LDH, ncol, kb, lhi), acc[i]);
          }
        }
#pragma unroll
        for (int i = 0; i < 4; ++i) {
          int col = (half * 4 + i) * 16 + llo;
#pragma unroll
          for (int r = 0; r < 8; ++r)
            Kb[(mt * 16 + r + 8 * lhi) * LDH + col] = (h16)acc[i][r];
        }
      }
      __syncthreads();
      // P += Qc @ Kc^T   (Kb rows are B^T rows)
      for (int ks = 0; ks < 4; ++ks) {
        int kb = ks * 32;
        v16h a = afrag(Qb, LDH, arow, kb, lhi);
#pragma unroll
        for (int j = 0; j < 2; ++j) {
          int nrow = (half * 2 + j) * 16 + llo;
          p[j] = wmma16(a, bfrag(Kb, LDH, nrow, kb, lhi), p[j]);
        }
      }
      // row sum-of-squares (from staged f16 values: self-consistent)
      if (tid < 128) {
        int row = tid & 63;
        const h16* s = (tid < 64) ? Qb : Kb;
        float acc = 0.f;
        for (int j = 0; j < 128; ++j) { float v = (float)s[row * LDH + j]; acc += v * v; }
        float* d = (tid < 64) ? qss : kss;
        if (cc == 0) d[row] = acc; else d[row] += acc;
      }
    }
    __syncthreads();
    // ---- finalize P -> Ps (f16) ----
    {
      float sh = expf(fminf(lscale[h], LOG100));
#pragma unroll
      for (int j = 0; j < 2; ++j) {
        int ncol = (half * 2 + j) * 16 + llo;
        float kinv = 1.f / fmaxf(sqrtf(kss[ncol]), 1e-12f);
#pragma unroll
        for (int r = 0; r < 8; ++r) {
          int row = mt * 16 + r + 8 * lhi;
          float qinv = 1.f / fmaxf(sqrtf(qss[row]), 1e-12f);
          Ps[row * LDV + ncol] = (h16)(p[j][r] * qinv * kinv * sh +
                                       Bias[((size_t)h * 64 + row) * 64 + ncol]);
        }
      }
    }

    // ---- out += (P @ V) @ WA over 3 chunks ----
    for (int cc = 0; cc < 3; ++cc) {
      const int colBase = h * HD + cc * 128;
      __syncthreads();
      // stage WV chunk
      {
        const uint4* s = (const uint4*)(WVt + (size_t)colBase * DIM);
        for (int v = tid; v < 2048; v += 256) {
          int r = v >> 4, kc = (v & 15) << 3;
          *(uint4*)&Wt[r * LDH + kc] = s[v];
        }
      }
      __syncthreads();
      {  // V chunk = Es @ WV + bv, stored transposed -> Vt[col][tok]
        v8f acc[4];
        for (int i = 0; i < 4; ++i) acc[i] = vzero8();
        for (int ks = 0; ks < 4; ++ks) {
          int kb = ks * 32;
          v16h a = afrag(Esh, LDH, arow, kb, lhi);
#pragma unroll
          for (int i = 0; i < 4; ++i) {
            int ncol = (half * 4 + i) * 16 + llo;
            acc[i] = wmma16(a, bfrag(Wt, LDH, ncol, kb, lhi), acc[i]);
          }
        }
#pragma unroll
        for (int i = 0; i < 4; ++i) {
          int col = (half * 4 + i) * 16 + llo;
          float vbias = WVb[colBase + col];
#pragma unroll
          for (int r = 0; r < 8; ++r)
            Vt[col * LDV + (mt * 16 + r + 8 * lhi)] = (h16)(acc[i][r] + vbias);
        }
      }
      __syncthreads();
      {  // Ac = Ps @ Vc  (K = 64)
        v8f acc[4];
        for (int i = 0; i < 4; ++i) acc[i] = vzero8();
        for (int ks = 0; ks < 2; ++ks) {
          int kb = ks * 32;
          v16h a = afrag(Ps, LDV, arow, kb, lhi);
#pragma unroll
          for (int i = 0; i < 4; ++i) {
            int ncol = (half * 4 + i) * 16 + llo;
            acc[i] = wmma16(a, bfrag(Vt, LDV, ncol, kb, lhi), acc[i]);
          }
        }
#pragma unroll
        for (int i = 0; i < 4; ++i) {
          int ncol = (half * 4 + i) * 16 + llo;
#pragma unroll
          for (int r = 0; r < 8; ++r)
            Ac[(mt * 16 + r + 8 * lhi) * LDH + ncol] = (h16)acc[i][r];
        }
      }
      __syncthreads();
      // stage WA chunk: Wt[n][k] = WAt[n][colBase + k]
      {
        for (int v = tid; v < 2048; v += 256) {
          int r = v >> 4, kc = (v & 15) << 3;
          *(uint4*)&Wt[r * LDH + kc] =
              *(const uint4*)&WAt[(size_t)r * HDT + colBase + kc];
        }
      }
      __syncthreads();
      {  // oacc += Ac @ WA[chunk rows, :]
        for (int ks = 0; ks < 4; ++ks) {
          int kb = ks * 32;
          v16h a = afrag(Ac, LDH, arow, kb, lhi);
#pragma unroll
          for (int i = 0; i < 4; ++i) {
            int ncol = (half * 4 + i) * 16 + llo;
            oacc[i] = wmma16(a, bfrag(Wt, LDH, ncol, kb, lhi), oacc[i]);
          }
        }
      }
    }
    __syncthreads();
  }

  // ---- write output (fp32) ----
  {
    float* ob = out + (size_t)bwin * (N_TOK * DIM);
#pragma unroll
    for (int i = 0; i < 4; ++i) {
      int ncol = (half * 4 + i) * 16 + llo;
      float bo = WAb[ncol];
#pragma unroll
      for (int r = 0; r < 8; ++r) {
        int row = mt * 16 + r + 8 * lhi;
        ob[row * DIM + ncol] = oacc[i][r] + bo;
      }
    }
  }
}

extern "C" void kernel_launch(void* const* d_in, const int* in_sizes, int n_in,
                              void* d_out, int out_size, void* d_ws, size_t ws_size,
                              hipStream_t stream) {
  const float* E   = (const float*)d_in[0];
  const float* rp  = (const float*)d_in[1];
  const int*   ri  = (const int*)d_in[2];
  const float* WQ  = (const float*)d_in[3];
  const float* WQb = (const float*)d_in[4];
  const float* WK  = (const float*)d_in[5];
  const float* WV  = (const float*)d_in[6];
  const float* WVb = (const float*)d_in[7];
  const float* WA  = (const float*)d_in[8];
  const float* WAb = (const float*)d_in[9];
  const float* w1  = (const float*)d_in[10];
  const float* b1  = (const float*)d_in[11];
  const float* w2  = (const float*)d_in[12];
  const float* ls  = (const float*)d_in[13];
  float* out = (float*)d_out;

  // workspace layout
  float* table = (float*)d_ws;                       // 900 floats
  float* Bias  = (float*)d_ws + 1024;                // 16384 floats
  char*  base  = (char*)d_ws + (1024 + 16384) * 4;   // 16B aligned
  const size_t WSZ = (size_t)DIM * HDT;              // 196608 elements
  h16* WQt = (h16*)base;
  h16* WKt = WQt + WSZ;
  h16* WVt = WKt + WSZ;
  h16* WAt = WVt + WSZ;

  bias_mlp_kernel<<<4, 256, 0, stream>>>(rp, w1, b1, w2, table);
  bias_expand_kernel<<<64, 256, 0, stream>>>(ri, table, Bias);
  // WQ/WK/WV: (K=128, N=1536) -> [1536][128] ; WA: (K=1536, N=128) -> [128][1536]
  transpose_conv_kernel<<<dim3(48, 4), 256, 0, stream>>>(WQ, WQt, 128, 1536);
  transpose_conv_kernel<<<dim3(48, 4), 256, 0, stream>>>(WK, WKt, 128, 1536);
  transpose_conv_kernel<<<dim3(48, 4), 256, 0, stream>>>(WV, WVt, 128, 1536);
  transpose_conv_kernel<<<dim3(4, 48), 256, 0, stream>>>(WA, WAt, 1536, 128);

  winmsa_kernel<<<B_WIN, 256, 0, stream>>>(E, WQt, WKt, WVt, WAt, WQb, WVb, WAb,
                                           ls, Bias, out);
}
